// GraphTransModel_43087111913536
// MI455X (gfx1250) — matrix operand order repaired and checked by reading, not compiled, not run on hardware
//
#include <hip/hip_runtime.h>
#include <hip/hip_bf16.h>
#include <math.h>

// ---------------- model constants (match reference) ----------------
#define NNODES  50000
#define IN_F    100
#define KPAD    128         // IN_F zero-padded so all GEMM K are multiples of 32
#define DMODEL  128
#define NHEADS  8
#define HDIM    16          // DMODEL / NHEADS
#define NLAYERS 3
#define NGRAPHS 64
#define FFDIM   512
#define NEDGES  800000

typedef __attribute__((ext_vector_type(16))) _Float16 v16h;
typedef __attribute__((ext_vector_type(4)))  _Float16 v4h;
typedef __attribute__((ext_vector_type(8)))  float    v8f;
typedef __attribute__((ext_vector_type(4)))  float    v4f;

// ---------------- helpers ----------------
__global__ void fill_kernel(float* __restrict__ p, float v, long n) {
    long i = (long)blockIdx.x * blockDim.x + threadIdx.x;
    if (i < n) p[i] = v;
}

// vectorized f32 -> f16 cast: 4 elements/thread (n must be a multiple of 4)
__global__ void cast_f16x4_kernel(const float* __restrict__ in, _Float16* __restrict__ out, long n4) {
    long i = (long)blockIdx.x * blockDim.x + threadIdx.x;
    if (i >= n4) return;
    *(v4h*)&out[i * 4] = __builtin_convertvector(*(const v4f*)&in[i * 4], v4h);
}

// rows of Kin f32 -> rows of Kout f16, zero-padded (for x: 100 -> 128)
__global__ void pad_cast_rows_kernel(const float* __restrict__ in, _Float16* __restrict__ out,
                                     int M, int Kin, int Kout) {
    long i = (long)blockIdx.x * blockDim.x + threadIdx.x;
    if (i >= (long)M * Kout) return;
    int m = (int)(i / Kout), k = (int)(i % Kout);
    out[i] = (k < Kin) ? (_Float16)in[(long)m * Kin + k] : (_Float16)0.0f;
}

// weight [K,N] f32 -> transposed [N,Kout] f16, zero-padded K (so GEMM B is row-major [N][K])
__global__ void transpose_cast_kernel(const float* __restrict__ in, _Float16* __restrict__ out,
                                      int K, int N, int Kout) {
    long i = (long)blockIdx.x * blockDim.x + threadIdx.x;
    if (i >= (long)N * Kout) return;
    int n = (int)(i / Kout), k = (int)(i % Kout);
    out[i] = (k < K) ? (_Float16)in[(long)k * N + n] : (_Float16)0.0f;
}

__device__ __forceinline__ void atomicMaxFloat(float* addr, float val) {
    int* ai = (int*)addr;
    int old = __float_as_int(*addr);
    while (__int_as_float(old) < val) {
        int assumed = old;
        old = atomicCAS(ai, assumed, __float_as_int(val));
        if (old == assumed) break;
    }
}

// generic-pointer low 32 bits == LDS byte offset
__device__ __forceinline__ unsigned lds_off(const void* p) {
    return (unsigned)(unsigned long long)(size_t)p;
}

// CDNA5 async copy: 16 bytes global -> LDS, tracked by ASYNCcnt (in-order completion)
__device__ __forceinline__ void async_copy_b128(unsigned ldsByteOff, const void* gptr) {
    asm volatile("global_load_async_to_lds_b128 %0, %1, off"
                 :: "v"(ldsByteOff), "v"((unsigned long long)(size_t)gptr)
                 : "memory");
}
__device__ __forceinline__ void wait_asynccnt0() {
    asm volatile("s_wait_asynccnt 0x0" ::: "memory");
}
// leave the 3 most-recently-issued (next-tile) copies in flight
__device__ __forceinline__ void wait_asynccnt3() {
    asm volatile("s_wait_asynccnt 0x3" ::: "memory");
}

// ---------------- WMMA GEMM ----------------
// C[M,N] = act( A[M,K] @ Bt[N,K]^T + bias[N] + res[M,N] ),  A/Bt are f16.
// Requirements: K % 32 == 0, N % 64 == 0 (M may be ragged).
// Block tile 128x64, K-step 32, 256 threads = 8 waves in a 4x2 grid,
// each wave computes a 32x32 quadrant = 4 x v_wmma_f32_16x16x32_f16.
// Double-buffered LDS; tiles prefetched with async b128 copies and pipelined
// against WMMA via the in-order ASYNCcnt (wait <=3 keeps next tile in flight).
#define BM 128
#define BN 64
#define BK 32
#define SA 40   // LDS row stride in halves: 80B = 16B-aligned, 20-bank stride (conflict-free)

__global__ __launch_bounds__(256) void gemm_wmma_kernel(
    const _Float16* __restrict__ A,   // [M,K]
    const _Float16* __restrict__ Bt,  // [N,K] (weights pre-transposed)
    const float* __restrict__ bias, const float* __restrict__ res,
    float* __restrict__ C, int M, int K, int N, int relu)
{
    __shared__ _Float16 Alds[2][BM * SA];   // [buf][row][k]
    __shared__ _Float16 Blds[2][BN * SA];   // [buf][n][k]

    const int tid   = threadIdx.x;
    const int lane  = tid & 31;
    const int wid   = tid >> 5;
    const int half_ = lane >> 4;     // 0: lanes 0-15, 1: lanes 16-31
    const int l16   = lane & 15;
    const int wm    = wid & 3;       // 4 waves along M
    const int wn    = wid >> 2;      // 2 waves along N
    const int rowBase = blockIdx.y * BM;
    const int colBase = blockIdx.x * BN;
    const bool fullTile = (rowBase + BM) <= M;   // block-uniform

    unsigned aBase[2] = { lds_off(&Alds[0][0]), lds_off(&Alds[1][0]) };
    unsigned bBase[2] = { lds_off(&Blds[0][0]), lds_off(&Blds[1][0]) };

    // per-thread staging slots: A gets 2 b128 chunks, B gets 1
    const int ar0 = tid >> 2,          ac0 = tid & 3;          // A chunk 0 (rows 0-63)
    const int ar1 = (tid + 256) >> 2,  ac1 = tid & 3;          // A chunk 1 (rows 64-127)
    const int br  = tid >> 2,          bc  = tid & 3;          // B chunk  (rows 0-63)

    auto stage_async = [&](int buf, int kbase) {
        async_copy_b128(aBase[buf] + (unsigned)(ar0 * SA + ac0 * 8) * 2,
                        A + (long)(rowBase + ar0) * K + kbase + ac0 * 8);
        async_copy_b128(aBase[buf] + (unsigned)(ar1 * SA + ac1 * 8) * 2,
                        A + (long)(rowBase + ar1) * K + kbase + ac1 * 8);
        async_copy_b128(bBase[buf] + (unsigned)(br * SA + bc * 8) * 2,
                        Bt + (long)(colBase + br) * K + kbase + bc * 8);
    };

    v8f acc[2][2];
    #pragma unroll
    for (int ti = 0; ti < 2; ++ti)
        #pragma unroll
        for (int tj = 0; tj < 2; ++tj)
            #pragma unroll
            for (int r = 0; r < 8; ++r)
                acc[ti][tj][r] = 0.0f;

    const int KT = K / BK;
    if (fullTile) stage_async(0, 0);   // prologue prefetch

    for (int kt = 0; kt < KT; ++kt) {
        const int cur = kt & 1;
        const int kbase = kt * BK;

        if (fullTile) {
            if (kt + 1 < KT) {
                stage_async(1 - cur, kbase + BK);  // prefetch next tile into other buffer
                wait_asynccnt3();                  // current tile done; next stays in flight
            } else {
                wait_asynccnt0();
            }
        } else {
            // ---- ragged M boundary block: guarded scalar staging (buffer `cur`) ----
            for (int idx = tid; idx < BM * BK; idx += 256) {
                int r = idx >> 5, c = idx & 31;
                int gr = rowBase + r;
                Alds[cur][r * SA + c] = (gr < M) ? A[(long)gr * K + kbase + c] : (_Float16)0.0f;
            }
            for (int idx = tid; idx < BN * BK; idx += 256) {
                int r = idx >> 5, c = idx & 31;   // r = n, c = k
                Blds[cur][r * SA + c] = Bt[(long)(colBase + r) * K + kbase + c];
            }
        }
        __syncthreads();

        // gather fragments per ISA 7.12.2 VGPR layouts
        v16h afrag[2], bfrag[2];
        #pragma unroll
        for (int ti = 0; ti < 2; ++ti) {
            const int r0 = wm * 32 + ti * 16 + l16;          // A row = lane%16
            unsigned int* au = (unsigned int*)&afrag[ti];
            #pragma unroll
            for (int vv = 0; vv < 8; ++vv) {
                // A 16x32: K = 16*(v>=4) + 8*half + 2*(v&3)  (2 halves/dword)
                int k0 = ((vv >> 2) << 4) + (half_ << 3) + ((vv & 3) << 1);
                au[vv] = *(const unsigned int*)&Alds[cur][r0 * SA + k0];
            }
        }
        #pragma unroll
        for (int tj = 0; tj < 2; ++tj) {
            const int c0 = wn * 32 + tj * 16 + l16;          // B col = lane%16
            unsigned int* bu = (unsigned int*)&bfrag[tj];
            #pragma unroll
            for (int vv = 0; vv < 8; ++vv) {
                // B 32x16: K = 16*(v>=4) + 8*half + 2*(v&3)  (2 halves/dword)
                int k0 = ((vv >> 2) << 4) + (half_ << 3) + ((vv & 3) << 1);
                bu[vv] = *(const unsigned int*)&Blds[cur][c0 * SA + k0];
            }
        }

        #pragma unroll
        for (int ti = 0; ti < 2; ++ti)
            #pragma unroll
            for (int tj = 0; tj < 2; ++tj)
                acc[ti][tj] = __builtin_amdgcn_wmma_f32_16x16x32_f16(
                    /*neg_a=*/false, afrag[ti], /*neg_b=*/false, bfrag[tj],
                    /*c_mod=*/(short)0, acc[ti][tj],
                    /*reuse_a=*/false, /*reuse_b=*/false);
        __syncthreads();   // waits DScnt: nobody still reads buf (1-cur) when it is re-staged
    }

    // epilogue: C/D layout -> M = vgpr + 8*half, N = lane%16
    #pragma unroll
    for (int ti = 0; ti < 2; ++ti) {
        #pragma unroll
        for (int tj = 0; tj < 2; ++tj) {
            const int gmBase = rowBase + wm * 32 + ti * 16 + half_ * 8;
            const int gn     = colBase + wn * 32 + tj * 16 + l16;
            #pragma unroll
            for (int r = 0; r < 8; ++r) {
                int gm = gmBase + r;
                if (gm < M && gn < N) {
                    float v = acc[ti][tj][r];
                    if (bias) v += bias[gn];
                    if (res)  v += res[(long)gm * N + gn];
                    if (relu) v = v > 0.0f ? v : 0.0f;
                    C[(long)gm * N + gn] = v;
                }
            }
        }
    }
}

// ---------------- edge (attention) kernels: 1 thread per (edge, head) ----------------
__global__ void edge_scores_kernel(const float* __restrict__ qk,
                                   const int* __restrict__ src,
                                   const int* __restrict__ dst,
                                   float* __restrict__ sc,
                                   float* __restrict__ mmax)
{
    long idx = (long)blockIdx.x * blockDim.x + threadIdx.x;
    if (idx >= (long)NEDGES * NHEADS) return;
    int e = (int)(idx / NHEADS), hh = (int)(idx % NHEADS);
    int s = src[e], d = dst[e];
    const float* q = qk + (long)s * (2 * DMODEL) + hh * HDIM;           // chunk 0 (source)
    const float* k = qk + (long)d * (2 * DMODEL) + DMODEL + hh * HDIM;  // chunk 1 (target)
    float a = 0.0f;
    #pragma unroll
    for (int i = 0; i < HDIM; ++i) a += q[i] * k[i];
    a *= 0.25f;                    // hd^{-1/2}, hd = 16
    sc[idx] = a;
    atomicMaxFloat(&mmax[(long)d * NHEADS + hh], a);
}

__global__ void edge_exp_kernel(float* __restrict__ sc,
                                const float* __restrict__ mmax,
                                const int* __restrict__ dst,
                                float* __restrict__ denom)
{
    long idx = (long)blockIdx.x * blockDim.x + threadIdx.x;
    if (idx >= (long)NEDGES * NHEADS) return;
    int e = (int)(idx / NHEADS), hh = (int)(idx % NHEADS);
    int d = dst[e];
    float ex = __expf(sc[idx] - mmax[(long)d * NHEADS + hh]);
    sc[idx] = ex;
    atomicAdd(&denom[(long)d * NHEADS + hh], ex);
}

__global__ void edge_aggregate_kernel(const float* __restrict__ sc,
                                      const float* __restrict__ denom,
                                      const float* __restrict__ vbuf,
                                      const int* __restrict__ src,
                                      const int* __restrict__ dst,
                                      float* __restrict__ aout)
{
    long idx = (long)blockIdx.x * blockDim.x + threadIdx.x;
    if (idx >= (long)NEDGES * NHEADS) return;
    int e = (int)(idx / NHEADS), hh = (int)(idx % NHEADS);
    int s = src[e], d = dst[e];
    float coeff = sc[idx] / (denom[(long)d * NHEADS + hh] + 1e-16f);
    const float* vv = vbuf + (long)s * DMODEL + hh * HDIM;
    float* o = aout + (long)d * DMODEL + hh * HDIM;
    #pragma unroll
    for (int i = 0; i < HDIM; ++i) atomicAdd(&o[i], coeff * vv[i]);
}

// ---------------- pooling ----------------
__global__ void pool_sum_kernel(const float* __restrict__ h,
                                const int* __restrict__ batch,
                                float* __restrict__ out,
                                float* __restrict__ cnt)
{
    long idx = (long)blockIdx.x * blockDim.x + threadIdx.x;
    if (idx >= (long)NNODES * DMODEL) return;
    int n = (int)(idx / DMODEL), d = (int)(idx % DMODEL);
    int b = batch[n];
    atomicAdd(&out[(long)b * DMODEL + d], h[idx]);
    if (d == 0) atomicAdd(&cnt[b], 1.0f);
}

__global__ void pool_div_kernel(float* __restrict__ out, const float* __restrict__ cnt) {
    int idx = blockIdx.x * blockDim.x + threadIdx.x;
    if (idx >= NGRAPHS * DMODEL) return;
    float c = cnt[idx / DMODEL];
    out[idx] /= (c > 1.0f ? c : 1.0f);
}

// ---------------- launch ----------------
static inline int cdiv(long a, long b) { return (int)((a + b - 1) / b); }

extern "C" void kernel_launch(void* const* d_in, const int* in_sizes, int n_in,
                              void* d_out, int out_size, void* d_ws, size_t ws_size,
                              hipStream_t stream) {
    const float* x     = (const float*)d_in[0];
    const float* emb_W = (const float*)d_in[1];
    const float* Wqk   = (const float*)d_in[2];
    const float* Wv    = (const float*)d_in[3];
    const float* Wo    = (const float*)d_in[4];
    const float* bo    = (const float*)d_in[5];
    const float* W1    = (const float*)d_in[6];
    const float* b1    = (const float*)d_in[7];
    const float* W2    = (const float*)d_in[8];
    const float* b2    = (const float*)d_in[9];
    const int* edge_index = (const int*)d_in[10];
    const int* batch      = (const int*)d_in[11];
    const int* src = edge_index;            // row 0
    const int* dst = edge_index + NEDGES;   // row 1
    float* out = (float*)d_out;

    // -------- workspace carve-out (f32 region first, then f16 region) --------
    float* ws   = (float*)d_ws;
    float* h    = ws; ws += (long)NNODES * DMODEL;
    float* qk   = ws; ws += (long)NNODES * 2 * DMODEL;
    float* vbuf = ws; ws += (long)NNODES * DMODEL;
    float* aout = ws; ws += (long)NNODES * DMODEL;
    float* sc   = ws; ws += (long)NEDGES * NHEADS;
    float* mmax = ws; ws += (long)NNODES * NHEADS;
    float* dn   = ws; ws += (long)NNODES * NHEADS;
    float* ffp  = ws; ws += (long)NNODES * FFDIM;
    float* cnt  = ws; ws += NGRAPHS;       // 64 floats keeps 16B alignment for f16 region

    _Float16* hws   = (_Float16*)ws;
    _Float16* act16 = hws; hws += (long)NNODES * FFDIM;                 // shared activation f16 buf
    _Float16* embT  = hws; hws += (long)DMODEL * KPAD;                  // [128][128] (K padded)
    _Float16* wqkT  = hws; hws += (long)NLAYERS * 2 * DMODEL * DMODEL;  // [256][128] per layer
    _Float16* wvT   = hws; hws += (long)NLAYERS * DMODEL * DMODEL;
    _Float16* woT   = hws; hws += (long)NLAYERS * DMODEL * DMODEL;
    _Float16* w1T   = hws; hws += (long)NLAYERS * FFDIM * DMODEL;       // [512][128]
    _Float16* w2T   = hws; hws += (long)NLAYERS * DMODEL * FFDIM;       // [128][512]

    auto gemm = [&](const _Float16* A, const _Float16* Bt, const float* bias,
                    const float* res, float* C, int M, int K, int N, int relu) {
        dim3 grid(cdiv(N, BN), cdiv(M, BM));
        gemm_wmma_kernel<<<grid, 256, 0, stream>>>(A, Bt, bias, res, C, M, K, N, relu);
    };
    auto cast = [&](const float* in, _Float16* o, long n) {
        cast_f16x4_kernel<<<cdiv(n / 4, 256), 256, 0, stream>>>(in, o, n / 4);
    };
    auto tcast = [&](const float* in, _Float16* o, int K, int N, int Kout) {
        transpose_cast_kernel<<<cdiv((long)N * Kout, 256), 256, 0, stream>>>(in, o, K, N, Kout);
    };

    const long EH = (long)NEDGES * NHEADS;
    const long ND = (long)NNODES * DMODEL;
    const long NH = (long)NNODES * NHEADS;

    // -------- one-time weight casts (transposed to [N][K] f16) --------
    tcast(emb_W, embT, IN_F, DMODEL, KPAD);
    for (int l = 0; l < NLAYERS; ++l) {
        tcast(Wqk + (long)l * DMODEL * 2 * DMODEL, wqkT + (long)l * 2 * DMODEL * DMODEL,
              DMODEL, 2 * DMODEL, DMODEL);
        tcast(Wv + (long)l * DMODEL * DMODEL, wvT + (long)l * DMODEL * DMODEL,
              DMODEL, DMODEL, DMODEL);
        tcast(Wo + (long)l * DMODEL * DMODEL, woT + (long)l * DMODEL * DMODEL,
              DMODEL, DMODEL, DMODEL);
        tcast(W1 + (long)l * DMODEL * FFDIM, w1T + (long)l * FFDIM * DMODEL,
              DMODEL, FFDIM, DMODEL);
        tcast(W2 + (long)l * FFDIM * DMODEL, w2T + (long)l * DMODEL * FFDIM,
              FFDIM, DMODEL, FFDIM);
    }

    // h = x @ emb_W   (x padded 100 -> 128 cols in f16)
    pad_cast_rows_kernel<<<cdiv((long)NNODES * KPAD, 256), 256, 0, stream>>>(
        x, act16, NNODES, IN_F, KPAD);
    gemm(act16, embT, nullptr, nullptr, h, NNODES, KPAD, DMODEL, 0);

    for (int l = 0; l < NLAYERS; ++l) {
        const _Float16* wqkT_l = wqkT + (long)l * 2 * DMODEL * DMODEL;
        const _Float16* wvT_l  = wvT  + (long)l * DMODEL * DMODEL;
        const _Float16* woT_l  = woT  + (long)l * DMODEL * DMODEL;
        const _Float16* w1T_l  = w1T  + (long)l * FFDIM * DMODEL;
        const _Float16* w2T_l  = w2T  + (long)l * DMODEL * FFDIM;
        const float* bo_l = bo + (long)l * DMODEL;
        const float* b1_l = b1 + (long)l * FFDIM;
        const float* b2_l = b2 + (long)l * DMODEL;

        // qk = h @ Wqk[l] ; v = h @ Wv[l]
        cast(h, act16, ND);
        gemm(act16, wqkT_l, nullptr, nullptr, qk, NNODES, DMODEL, 2 * DMODEL, 0);
        gemm(act16, wvT_l,  nullptr, nullptr, vbuf, NNODES, DMODEL, DMODEL, 0);

        // segment softmax + scatter aggregation
        fill_kernel<<<cdiv(NH, 256), 256, 0, stream>>>(mmax, -1e30f, NH);
        fill_kernel<<<cdiv(NH, 256), 256, 0, stream>>>(dn,    0.0f,  NH);
        fill_kernel<<<cdiv(ND, 256), 256, 0, stream>>>(aout,  0.0f,  ND);
        edge_scores_kernel   <<<cdiv(EH, 256), 256, 0, stream>>>(qk, src, dst, sc, mmax);
        edge_exp_kernel      <<<cdiv(EH, 256), 256, 0, stream>>>(sc, mmax, dst, dn);
        edge_aggregate_kernel<<<cdiv(EH, 256), 256, 0, stream>>>(sc, dn, vbuf, src, dst, aout);

        // h = h + aout @ Wo[l] + bo[l]   (residual fused, element-safe in-place)
        cast(aout, act16, ND);
        gemm(act16, woT_l, bo_l, h, h, NNODES, DMODEL, DMODEL, 0);
        // ffp = relu(h @ W1[l] + b1[l])
        cast(h, act16, ND);
        gemm(act16, w1T_l, b1_l, nullptr, ffp, NNODES, DMODEL, FFDIM, 1);
        // h = h + ffp @ W2[l] + b2[l]
        cast(ffp, act16, (long)NNODES * FFDIM);
        gemm(act16, w2T_l, b2_l, h, h, NNODES, FFDIM, DMODEL, 0);
    }

    // global mean pool per graph
    fill_kernel<<<cdiv(NGRAPHS * DMODEL, 256), 256, 0, stream>>>(out, 0.0f, NGRAPHS * DMODEL);
    fill_kernel<<<1, 256, 0, stream>>>(cnt, 0.0f, NGRAPHS);
    pool_sum_kernel<<<cdiv(ND, 256), 256, 0, stream>>>(h, batch, out, cnt);
    pool_div_kernel<<<cdiv(NGRAPHS * DMODEL, 256), 256, 0, stream>>>(out, cnt);
}
